// MultiHeadAttention_14963666059752
// MI455X (gfx1250) — matrix operand order repaired and checked by reading, not compiled, not run on hardware
//
#include <hip/hip_runtime.h>
#include <stdint.h>

#define EMB      1024
#define HEADS    16
#define HDIM     64
#define BATCH    4
#define SEQ      2048
#define TOKENS   (BATCH * SEQ)        // 8192
#define QKV_N    (3 * EMB)            // 3072
#define BH       (BATCH * HEADS)      // 64
#define SEGELTS  ((size_t)BH * SEQ * HDIM)   // 8388608 elements per [B,H,N,D] segment

typedef unsigned short u16;
typedef __bf16 bf16_t;
typedef __attribute__((ext_vector_type(16))) bf16_t bvec16;
typedef __attribute__((ext_vector_type(8)))  float  v8f;

union BFrag {
    bvec16 v;
    uint4  q[2];
    u16    h[16];
};

__device__ __forceinline__ u16 f2bf(float f) {
    __bf16 h = (__bf16)f;                      // native RNE convert
    return __builtin_bit_cast(u16, h);
}

__device__ __forceinline__ v8f wmma_bf16(const BFrag& a, const BFrag& b, v8f c) {
    return __builtin_amdgcn_wmma_f32_16x16x32_bf16(false, a.v, false, b.v,
                                                   (short)0, c, false, false);
}

// A-operand frag (16x32, K-major rows): two 8-elem runs at half*8 and 16+half*8
__device__ __forceinline__ void load_a_bf16(BFrag& f, const u16* rowk, int half) {
    const u16* p = rowk + half * 8;
    f.q[0] = *(const uint4*)(p);
    f.q[1] = *(const uint4*)(p + 16);
}

// B-operand frag (32x16, K-major rows per column n): two 8-elem runs at half*8 and 16+half*8
__device__ __forceinline__ void load_b_bf16(BFrag& f, const u16* rowk, int half) {
    const u16* p = rowk + half * 8;
    f.q[0] = *(const uint4*)(p);
    f.q[1] = *(const uint4*)(p + 16);
}

// Async DMA of one 16B chunk: global -> LDS (ASYNCcnt-tracked)
__device__ __forceinline__ void async_b128(uint32_t lds_byte_addr, const u16* g) {
    asm volatile("global_load_async_to_lds_b128 %0, %1, off"
                 :: "v"(lds_byte_addr), "v"(g) : "memory");
}
__device__ __forceinline__ void wait_async(int n) {
    if (n == 0) asm volatile("s_wait_asynccnt 0x0" ::: "memory");
    else        asm volatile("s_wait_asynccnt 0x4" ::: "memory");
}
__device__ __forceinline__ uint32_t lds_addr(const void* p) {
    return (uint32_t)(uintptr_t)p;             // low 32 bits = LDS byte offset
}

// Stage a 128-row x 64-elem bf16 tile (row pad -> 72 elems / 144B) via async DMA.
__device__ __forceinline__ void stage128x64_async(uint32_t ldsBase, const u16* gsrc,
                                                  size_t gRowElems, int tid) {
    #pragma unroll
    for (int it = 0; it < 4; ++it) {
        int idx = tid + it * 256;              // 1024 chunks of 16B
        int r = idx >> 3, c = idx & 7;
        async_b128(ldsBase + r * 144 + c * 16, gsrc + (size_t)r * gRowElems + c * 8);
    }
}

// ---------------------------------------------------------------------------
// Pre-pass: convert x (fp32) -> bf16, flat copy. 8 elems / thread.
// ---------------------------------------------------------------------------
__global__ __launch_bounds__(256)
void cvt_x_kernel(const float* __restrict__ x, u16* __restrict__ xbf)
{
    size_t i = ((size_t)blockIdx.x * 256 + threadIdx.x) * 8;
    float4 a = *(const float4*)(x + i);
    float4 b = *(const float4*)(x + i + 4);
    union { uint4 q; u16 h[8]; } o;
    o.h[0] = f2bf(a.x); o.h[1] = f2bf(a.y); o.h[2] = f2bf(a.z); o.h[3] = f2bf(a.w);
    o.h[4] = f2bf(b.x); o.h[5] = f2bf(b.y); o.h[6] = f2bf(b.z); o.h[7] = f2bf(b.w);
    *(uint4*)(xbf + i) = o.q;
}

// ---------------------------------------------------------------------------
// Pre-pass: W [K][N] fp32 -> WT [N][K] bf16 (K-major for the B operand).
// 32x32 tiles through LDS. Grid (N/32, K/32), 256 threads (32x8).
// ---------------------------------------------------------------------------
__global__ __launch_bounds__(256)
void tcvt_kernel(const float* __restrict__ W, u16* __restrict__ WT, int K, int N)
{
    __shared__ u16 tile[32][34];
    const int tx = threadIdx.x & 31, ty = threadIdx.x >> 5;
    const int n0 = blockIdx.x * 32, k0 = blockIdx.y * 32;
    #pragma unroll
    for (int r = 0; r < 4; ++r) {
        int row = ty + r * 8;
        tile[row][tx] = f2bf(W[(size_t)(k0 + row) * N + n0 + tx]);
    }
    __syncthreads();
    #pragma unroll
    for (int r = 0; r < 4; ++r) {
        int row = ty + r * 8;
        WT[(size_t)(n0 + row) * K + k0 + tx] = tile[tx][row];
    }
}

// ---------------------------------------------------------------------------
// Kernel 1: qkv = xbf @ WqkvT^T + bqkv ; scatter Q/K [B,H,N,D], V [B,H,D,N] bf16.
// Tile 128x128, BK=64, double-buffered async B staging. Grid (64, 24).
// ---------------------------------------------------------------------------
__global__ __launch_bounds__(256)
void qkv_gemm_kernel(const u16* __restrict__ xbf, const u16* __restrict__ wT,
                     const float* __restrict__ bqkv,
                     u16* __restrict__ qw, u16* __restrict__ kw, u16* __restrict__ vt)
{
    __shared__ u16 Bs[2][128][72];             // 2 x 18KB

    const int tid  = threadIdx.x;
    const int lane = tid & 31, w = tid >> 5;
    const int half = lane >> 4, ln = lane & 15;
    const int wy = w & 3, wx = w >> 2;
    const int m0 = blockIdx.x * 128;
    const int n0 = blockIdx.y * 128;
    const uint32_t bsBase = lds_addr(&Bs[0][0][0]);

    v8f acc[2][4] = {};

    stage128x64_async(bsBase, wT + (size_t)n0 * EMB, EMB, tid);

    for (int ti = 0; ti < EMB / 64; ++ti) {
        const int cur = ti & 1;
        wait_async(0);
        __syncthreads();                       // tile ti landed (all waves)
        if (ti + 1 < EMB / 64)
            stage128x64_async(bsBase + (cur ^ 1) * (128 * 144),
                              wT + (size_t)n0 * EMB + (ti + 1) * 64, EMB, tid);
        #pragma unroll
        for (int ks = 0; ks < 2; ++ks) {
            BFrag a0, a1;
            const u16* arow0 = xbf + (size_t)(m0 + wy * 32 + ln) * EMB + ti * 64 + ks * 32;
            load_a_bf16(a0, arow0, half);
            load_a_bf16(a1, arow0 + (size_t)16 * EMB, half);
            #pragma unroll
            for (int nt = 0; nt < 4; ++nt) {
                BFrag b;
                load_b_bf16(b, &Bs[cur][wx * 64 + nt * 16 + ln][ks * 32], half);
                acc[0][nt] = wmma_bf16(a0, b, acc[0][nt]);
                acc[1][nt] = wmma_bf16(a1, b, acc[1][nt]);
            }
        }
    }

    // epilogue: bias, fold softmax scale into Q, scatter (sel uniform per block)
    const int n_blk = n0;                      // block spans one sel entirely
    const int sel = n_blk >> 10;
    #pragma unroll
    for (int mi = 0; mi < 2; ++mi) {
        int tokbase = m0 + wy * 32 + mi * 16 + 8 * half;
        int bb = tokbase >> 11, ntok0 = tokbase & 2047;
        #pragma unroll
        for (int nt = 0; nt < 4; ++nt) {
            int n = n0 + wx * 64 + nt * 16 + ln;
            float bias = bqkv[n];
            int r = n & 1023;
            int hh = r >> 6, d = r & 63;
            if (sel == 2) {                    // V: [B,H,D,N] -> 8 consecutive tokens
                union { uint4 q; u16 h[8]; } o;
                #pragma unroll
                for (int v = 0; v < 8; ++v)
                    o.h[v] = f2bf(acc[mi][nt][v] + bias);
                *(uint4*)(vt + ((size_t)(bb * HEADS + hh) * HDIM + d) * SEQ + ntok0) = o.q;
            } else {
                float scale = (sel == 0) ? 0.125f : 1.0f;
                u16* dst = (sel == 0) ? qw : kw;
                #pragma unroll
                for (int v = 0; v < 8; ++v)
                    dst[(((size_t)(bb * HEADS + hh)) * SEQ + ntok0 + v) * HDIM + d] =
                        f2bf((acc[mi][nt][v] + bias) * scale);
            }
        }
    }
}

// ---------------------------------------------------------------------------
// Kernel 2: flash attention per (b,h). 128 query rows/block, 8 waves x 16 rows.
// K and V tiles staged via async DMA; S compute overlaps V copy. Grid (16, 64).
// ---------------------------------------------------------------------------
__global__ __launch_bounds__(256)
void attn_kernel(const u16* __restrict__ qw, const u16* __restrict__ kw,
                 const u16* __restrict__ vt, u16* __restrict__ ctx)
{
    __shared__ u16 Ks[128][72];      // K tile [key][d]   18 KB
    __shared__ u16 Vt[64][136];      // V^T tile [d][key] 17 KB
    __shared__ u16 Ps[8][16][48];    // per-wave P chunk  12 KB

    const int tid  = threadIdx.x;
    const int lane = tid & 31, w = tid >> 5;
    const int half = lane >> 4, ln = lane & 15;
    const int bh = blockIdx.y;
    const int qbase = blockIdx.x * 128 + w * 16;
    const size_t hb = (size_t)bh * SEQ;
    const uint32_t ksBase = lds_addr(&Ks[0][0]);
    const uint32_t vtBase = lds_addr(&Vt[0][0]);

    BFrag qf[2];
    #pragma unroll
    for (int ks = 0; ks < 2; ++ks)
        load_a_bf16(qf[ks], qw + (hb + qbase + ln) * HDIM + ks * 32, half);

    v8f o[4] = {};
    float mrow[8], lrow[8];
    #pragma unroll
    for (int v = 0; v < 8; ++v) { mrow[v] = -1e30f; lrow[v] = 0.f; }

    for (int kt = 0; kt < SEQ / 128; ++kt) {
        // async-stage K tile (4 chunks/thread), then V tile (4 chunks/thread)
        stage128x64_async(ksBase, kw + (hb + kt * 128) * HDIM, HDIM, tid);
        #pragma unroll
        for (int it = 0; it < 4; ++it) {       // V: 64 rows x 256B, pad rows 272B
            int idx = tid + it * 256;
            int d = idx >> 4, c = idx & 15;
            async_b128(vtBase + d * 272 + c * 16,
                       vt + ((size_t)bh * HDIM + d) * SEQ + kt * 128 + c * 8);
        }
        wait_async(4);                         // K portion landed (V still in flight)
        __syncthreads();

        // S = Q @ K^T
        v8f s[8] = {};
        #pragma unroll
        for (int ks = 0; ks < 2; ++ks) {
            #pragma unroll
            for (int nt = 0; nt < 8; ++nt) {
                BFrag b;
                load_b_bf16(b, &Ks[nt * 16 + ln][ks * 32], half);
                s[nt] = wmma_bf16(qf[ks], b, s[nt]);
            }
        }

        // online softmax stats (shfl within 16-lane halves; wave32)
        float mnew[8], alpha[8], rs[8];
        #pragma unroll
        for (int v = 0; v < 8; ++v) {
            float mx = mrow[v];
            #pragma unroll
            for (int nt = 0; nt < 8; ++nt) mx = fmaxf(mx, s[nt][v]);
            #pragma unroll
            for (int msk = 1; msk <= 8; msk <<= 1)
                mx = fmaxf(mx, __shfl_xor(mx, msk, 32));
            mnew[v]  = mx;
            alpha[v] = __expf(mrow[v] - mx);
            mrow[v]  = mx;
            rs[v]    = 0.f;
        }
        #pragma unroll
        for (int dt = 0; dt < 4; ++dt)
            #pragma unroll
            for (int v = 0; v < 8; ++v) o[dt][v] *= alpha[v];

        wait_async(0);                         // V tile landed
        __syncthreads();

        // P = exp(S - m) -> chunked LDS transpose -> O += P @ V
        #pragma unroll
        for (int ks2 = 0; ks2 < 4; ++ks2) {
            asm volatile("s_wait_dscnt 0" ::: "memory");
            #pragma unroll
            for (int j = 0; j < 2; ++j) {
                int nt = ks2 * 2 + j;
                #pragma unroll
                for (int v = 0; v < 8; ++v) {
                    float p = __expf(s[nt][v] - mnew[v]);
                    rs[v] += p;
                    Ps[w][8 * half + v][j * 16 + ln] = f2bf(p);
                }
            }
            asm volatile("s_wait_dscnt 0" ::: "memory");
            BFrag a;
            load_a_bf16(a, &Ps[w][ln][0], half);
            #pragma unroll
            for (int dt = 0; dt < 4; ++dt) {
                BFrag b;
                load_b_bf16(b, &Vt[dt * 16 + ln][ks2 * 32], half);
                o[dt] = wmma_bf16(a, b, o[dt]);
            }
        }

        #pragma unroll
        for (int v = 0; v < 8; ++v) {
            float t = rs[v];
            #pragma unroll
            for (int msk = 1; msk <= 8; msk <<= 1)
                t += __shfl_xor(t, msk, 32);
            lrow[v] = lrow[v] * alpha[v] + t;
        }
        __syncthreads();                       // reads done before next tile's DMA
    }

    const int bb = bh >> 4, hh = bh & 15;
    #pragma unroll
    for (int dt = 0; dt < 4; ++dt) {
        #pragma unroll
        for (int v = 0; v < 8; ++v) {
            int row = qbase + 8 * half + v;
            int d = dt * 16 + ln;
            ctx[((size_t)(bb * SEQ + row)) * EMB + hh * HDIM + d] =
                f2bf(o[dt][v] / lrow[v]);
        }
    }
}

// ---------------------------------------------------------------------------
// Kernel 3: out = ctx @ WoutT^T + bout (fp32). Same structure as kernel 1.
// ---------------------------------------------------------------------------
__global__ __launch_bounds__(256)
void out_gemm_kernel(const u16* __restrict__ ctx, const u16* __restrict__ wT,
                     const float* __restrict__ bout, float* __restrict__ out)
{
    __shared__ u16 Bs[2][128][72];

    const int tid  = threadIdx.x;
    const int lane = tid & 31, w = tid >> 5;
    const int half = lane >> 4, ln = lane & 15;
    const int wy = w & 3, wx = w >> 2;
    const int m0 = blockIdx.x * 128;
    const int n0 = blockIdx.y * 128;
    const uint32_t bsBase = lds_addr(&Bs[0][0][0]);

    v8f acc[2][4] = {};

    stage128x64_async(bsBase, wT + (size_t)n0 * EMB, EMB, tid);

    for (int ti = 0; ti < EMB / 64; ++ti) {
        const int cur = ti & 1;
        wait_async(0);
        __syncthreads();
        if (ti + 1 < EMB / 64)
            stage128x64_async(bsBase + (cur ^ 1) * (128 * 144),
                              wT + (size_t)n0 * EMB + (ti + 1) * 64, EMB, tid);
        #pragma unroll
        for (int ks = 0; ks < 2; ++ks) {
            BFrag a0, a1;
            const u16* arow0 = ctx + (size_t)(m0 + wy * 32 + ln) * EMB + ti * 64 + ks * 32;
            load_a_bf16(a0, arow0, half);
            load_a_bf16(a1, arow0 + (size_t)16 * EMB, half);
            #pragma unroll
            for (int nt = 0; nt < 4; ++nt) {
                BFrag b;
                load_b_bf16(b, &Bs[cur][wx * 64 + nt * 16 + ln][ks * 32], half);
                acc[0][nt] = wmma_bf16(a0, b, acc[0][nt]);
                acc[1][nt] = wmma_bf16(a1, b, acc[1][nt]);
            }
        }
    }

    #pragma unroll
    for (int mi = 0; mi < 2; ++mi) {
        int tokbase = m0 + wy * 32 + mi * 16 + 8 * half;
        #pragma unroll
        for (int nt = 0; nt < 4; ++nt) {
            int n = n0 + wx * 64 + nt * 16 + ln;
            float bias = bout[n];
            #pragma unroll
            for (int v = 0; v < 8; ++v)
                out[(size_t)(tokbase + v) * EMB + n] = acc[mi][nt][v] + bias;
        }
    }
}

// ---------------------------------------------------------------------------
extern "C" void kernel_launch(void* const* d_in, const int* in_sizes, int n_in,
                              void* d_out, int out_size, void* d_ws, size_t ws_size,
                              hipStream_t stream)
{
    const float* x    = (const float*)d_in[0];
    const float* Wqkv = (const float*)d_in[1];
    const float* bqkv = (const float*)d_in[2];
    const float* Wout = (const float*)d_in[3];
    const float* bout = (const float*)d_in[4];
    float* out = (float*)d_out;

    u16* qw    = (u16*)d_ws;               // [B,H,N,D] bf16, Q pre-scaled
    u16* kw    = qw  + SEGELTS;            // [B,H,N,D] bf16
    u16* vt    = kw  + SEGELTS;            // [B,H,D,N] bf16 (pre-transposed V)
    u16* ctx   = vt  + SEGELTS;            // [B*N, EMB] bf16
    u16* xbf   = ctx + SEGELTS;            // [B*N, EMB] bf16
    u16* wqkvT = xbf + SEGELTS;            // [3072, 1024] bf16 (W^T, K-major)
    u16* woutT = wqkvT + (size_t)QKV_N * EMB;  // [1024, 1024] bf16

    cvt_x_kernel<<<(TOKENS * EMB) / (256 * 8), 256, 0, stream>>>(x, xbf);
    tcvt_kernel<<<dim3(QKV_N / 32, EMB / 32), 256, 0, stream>>>(Wqkv, wqkvT, EMB, QKV_N);
    tcvt_kernel<<<dim3(EMB / 32, EMB / 32), 256, 0, stream>>>(Wout, woutT, EMB, EMB);

    qkv_gemm_kernel<<<dim3(TOKENS / 128, QKV_N / 128), 256, 0, stream>>>(
        xbf, wqkvT, bqkv, qw, kw, vt);
    attn_kernel<<<dim3(SEQ / 128, BH), 256, 0, stream>>>(qw, kw, vt, ctx);
    out_gemm_kernel<<<dim3(TOKENS / 128, EMB / 128), 256, 0, stream>>>(
        ctx, woutT, bout, out);
}
